// MLFPN_GCN_2405181685967
// MI455X (gfx1250) — matrix operand order repaired and checked by reading, not compiled, not run on hardware
//
#include <hip/hip_runtime.h>
#include <stdint.h>

#define NN 50000
#define NE 1600000

typedef __attribute__((ext_vector_type(16))) __bf16 v16bf;
typedef __attribute__((ext_vector_type(8)))  float  v8f;

__device__ __forceinline__ unsigned short f32_to_bf16_rne(float f) {
  unsigned int u = __float_as_uint(f);
  unsigned int r = u + 0x7FFFu + ((u >> 16) & 1u);
  return (unsigned short)(r >> 16);
}

// ---- CDNA5 async global->LDS copy (ASYNCcnt-tracked, bypasses VGPRs) ----
__device__ __forceinline__ void async_ld16(unsigned loff, const void* gsrc) {
  unsigned long long ga = (unsigned long long)(uintptr_t)gsrc;
  asm volatile("global_load_async_to_lds_b128 %0, %1, off"
               :: "v"(loff), "v"(ga)
               : "memory");
}
__device__ __forceinline__ unsigned lds_off(const void* p) {
  // Generic LDS pointer: addr[31:0] is the LDS byte address (ISA §10.2 aperture).
  return (unsigned)(uintptr_t)p;
}

__device__ __forceinline__ void wait_async4() {
#if __has_builtin(__builtin_amdgcn_s_wait_asynccnt)
  __builtin_amdgcn_s_wait_asynccnt(4);
#else
  asm volatile("s_wait_asynccnt 0x4" ::: "memory");
#endif
}
__device__ __forceinline__ void wait_async0() {
#if __has_builtin(__builtin_amdgcn_s_wait_asynccnt)
  __builtin_amdgcn_s_wait_asynccnt(0);
#else
  asm volatile("s_wait_asynccnt 0x0" ::: "memory");
#endif
}

// ---------------------------------------------------------------- converts
__global__ void cvt_bf16_relu(const float* __restrict__ x,
                              unsigned short* __restrict__ y,
                              long long n, int relu) {
  long long i = (long long)blockIdx.x * blockDim.x + threadIdx.x;
  long long stride = (long long)gridDim.x * blockDim.x;
  for (; i < n; i += stride) {
    float v = x[i];
    if (relu) v = fmaxf(v, 0.0f);
    y[i] = f32_to_bf16_rne(v);
  }
}

// W [K,N] f32 row-major -> Wt [N,K] bf16 row-major (i.e. column-major W)
__global__ void cvt_w_transpose(const float* __restrict__ W,
                                unsigned short* __restrict__ Wt,
                                int K, int N) {
  long long total = (long long)K * N;
  long long i = (long long)blockIdx.x * blockDim.x + threadIdx.x;
  long long stride = (long long)gridDim.x * blockDim.x;
  for (; i < total; i += stride) {
    int k = (int)(i / N);
    int n = (int)(i % N);
    Wt[(size_t)n * K + k] = f32_to_bf16_rne(W[i]);
  }
}

// ---------------------------------------------------------------- GEMM
// C[M,N] f32 = A[M,K](bf16) @ Bt[N,K](bf16)^T + bias[N]
#define BM 128
#define BN 128
#define BK 32
#define LDT 40  // padded LDS row stride in elements (80 bytes, 16B aligned)

__launch_bounds__(256)
__global__ void gcn_gemm_bf16(const unsigned short* __restrict__ A,
                              const unsigned short* __restrict__ Bt,
                              const float* __restrict__ bias,
                              float* __restrict__ C,
                              int M, int K, int N) {
  __shared__ __align__(16) unsigned short As[2][BM * LDT];
  __shared__ __align__(16) unsigned short Bs[2][BN * LDT];

  const int tid  = threadIdx.x;
  const int lane = tid & 31;
  const int wave = tid >> 5;
  const int half = lane >> 4;   // 0: lanes 0-15, 1: lanes 16-31
  const int l16  = lane & 15;
  const int wm   = wave & 3;    // 4 wave-strips in M (32 rows each)
  const int wn   = wave >> 2;   // 2 wave-strips in N (64 cols each)
  const int bm0  = blockIdx.x * BM;
  const int bn0  = blockIdx.y * BN;

  const v8f vzero = {0.f, 0.f, 0.f, 0.f, 0.f, 0.f, 0.f, 0.f};
  v8f acc[2][4];
#pragma unroll
  for (int mi = 0; mi < 2; mi++)
#pragma unroll
    for (int ni = 0; ni < 4; ni++) acc[mi][ni] = vzero;

  union Frag { v16bf v; uint4 q[2]; };

  // ---- per-thread copy assignment: tile = 512 x 16B chunks, 2 per thread.
  // chunk0 = tid -> (row r0, 16B-col c0); chunk1 = tid+256 -> (row r0+64, c0)
  const int r0 = tid >> 2;
  const int c0 = tid & 3;
  int growA0 = bm0 + r0;       if (growA0 > M - 1) growA0 = M - 1;
  int growA1 = bm0 + r0 + 64;  if (growA1 > M - 1) growA1 = M - 1;
  const unsigned short* gA0 = A  + (size_t)growA0 * K + c0 * 8;
  const unsigned short* gA1 = A  + (size_t)growA1 * K + c0 * 8;
  const unsigned short* gB0 = Bt + (size_t)(bn0 + r0) * K + c0 * 8;
  const unsigned short* gB1 = Bt + (size_t)(bn0 + r0 + 64) * K + c0 * 8;
  unsigned lA0[2], lA1[2], lB0[2], lB1[2];
#pragma unroll
  for (int b = 0; b < 2; b++) {
    lA0[b] = lds_off(&As[b][r0 * LDT + c0 * 8]);
    lA1[b] = lds_off(&As[b][(r0 + 64) * LDT + c0 * 8]);
    lB0[b] = lds_off(&Bs[b][r0 * LDT + c0 * 8]);
    lB1[b] = lds_off(&Bs[b][(r0 + 64) * LDT + c0 * 8]);
  }

  const int ntile = K / BK;

  // ---- prologue: issue tile 0 into buffer 0 (4 async loads per thread)
  async_ld16(lA0[0], gA0);
  async_ld16(lA1[0], gA1);
  async_ld16(lB0[0], gB0);
  async_ld16(lB1[0], gB1);
  gA0 += BK; gA1 += BK; gB0 += BK; gB1 += BK;

  for (int t = 0; t < ntile; t++) {
    const int cur = t & 1;
    // ---- issue next tile into the other buffer, then wait for current tile
    if (t + 1 < ntile) {
      const int nb = cur ^ 1;
      async_ld16(lA0[nb], gA0);
      async_ld16(lA1[nb], gA1);
      async_ld16(lB0[nb], gB0);
      async_ld16(lB1[nb], gB1);
      gA0 += BK; gA1 += BK; gB0 += BK; gB1 += BK;
      wait_async4();  // in-order completion => current tile's 4 loads are done
    } else {
      wait_async0();
    }
    __syncthreads();

    // ---- fragment loads (per ISA 16-bit A/B VGPR layouts)
    Frag a[2], b[4];
    const int akoff = half ? 8 : 0;   // A: lanes<16 -> K0..7,K16..23 ; lanes>=16 -> K8..15,K24..31
#pragma unroll
    for (int mi = 0; mi < 2; mi++) {
      const unsigned short* p = &As[cur][(wm * 32 + mi * 16 + l16) * LDT];
      a[mi].q[0] = *(const uint4*)(p + akoff);
      a[mi].q[1] = *(const uint4*)(p + akoff + 16);
    }
    const int bkoff = half ? 8 : 0;   // B: lanes<16 -> K0..7,K16..23 ; lanes>=16 -> K8..15,K24..31
#pragma unroll
    for (int ni = 0; ni < 4; ni++) {
      const unsigned short* p = &Bs[cur][(wn * 64 + ni * 16 + l16) * LDT];
      b[ni].q[0] = *(const uint4*)(p + bkoff);
      b[ni].q[1] = *(const uint4*)(p + bkoff + 16);
    }

#pragma unroll
    for (int mi = 0; mi < 2; mi++)
#pragma unroll
      for (int ni = 0; ni < 4; ni++)
        acc[mi][ni] = __builtin_amdgcn_wmma_f32_16x16x32_bf16(
            false, a[mi].v, false, b[ni].v, (short)0, acc[mi][ni], false, false);

    __syncthreads();  // protect buffer `cur` before next iteration overwrites it
  }

  // ---- epilogue: C/D layout: VGPR r -> row base+r (lanes 0-15) / base+8+r (lanes 16-31)
#pragma unroll
  for (int ni = 0; ni < 4; ni++) {
    int col = bn0 + wn * 64 + ni * 16 + l16;
    float bv = bias[col];
#pragma unroll
    for (int mi = 0; mi < 2; mi++) {
      int rbase = bm0 + wm * 32 + mi * 16 + half * 8;
#pragma unroll
      for (int r = 0; r < 8; r++) {
        int row = rbase + r;
        if (row < M) C[(size_t)row * N + col] = acc[mi][ni][r] + bv;
      }
    }
  }
}

// ---------------------------------------------------------------- SpMM
// H[row] += val * S[col]  (COO scatter, one block per edge)
__global__ void gcn_spmm(const float* __restrict__ S,
                         const long long* __restrict__ ei,
                         const float* __restrict__ ev,
                         float* __restrict__ H, int dout) {
  int e = blockIdx.x;
  long long r = ei[e];
  long long c = ei[(long long)NE + e];
  float v = ev[e];
  const float4* src = (const float4*)(S + (size_t)c * dout);
  float* dst = H + (size_t)r * dout;
  int n4 = dout >> 2;
  for (int i = threadIdx.x; i < n4; i += blockDim.x) {
    float4 m = src[i];
    unsafeAtomicAdd(dst + 4 * i + 0, v * m.x);
    unsafeAtomicAdd(dst + 4 * i + 1, v * m.y);
    unsafeAtomicAdd(dst + 4 * i + 2, v * m.z);
    unsafeAtomicAdd(dst + 4 * i + 3, v * m.w);
  }
}

// ---------------------------------------------------------------- launch
extern "C" void kernel_launch(void* const* d_in, const int* in_sizes, int n_in,
                              void* d_out, int out_size, void* d_ws, size_t ws_size,
                              hipStream_t stream) {
  const float*     fea = (const float*)d_in[0];
  const long long* ei  = (const long long*)d_in[1];  // int64 [2, NE]
  const float*     ev  = (const float*)d_in[2];
  const float*     W1  = (const float*)d_in[3];
  const float*     b1  = (const float*)d_in[4];
  const float*     W2  = (const float*)d_in[5];
  const float*     b2  = (const float*)d_in[6];
  const float*     W3  = (const float*)d_in[7];
  const float*     b3  = (const float*)d_in[8];
  float*           out = (float*)d_out;

  char* ws = (char*)d_ws;
  unsigned short* Xb = (unsigned short*)(ws);                   // 50000*2048*2 = 204,800,000 B
  unsigned short* Wt = (unsigned short*)(ws + 204800000LL);     // up to 2048*2048*2 = 8,388,608 B
  float*          S  = (float*)(ws + 213188608LL);              // 50000*2048*4 = 409,600,000 B
  float*          H  = (float*)(ws + 622788608LL);              // 50000*2048*4 = 409,600,000 B

  // ---------------- layer 1: 1024 -> 2048
  cvt_bf16_relu<<<4096, 256, 0, stream>>>(fea, Xb, 50000LL * 1024, 0);
  cvt_w_transpose<<<4096, 256, 0, stream>>>(W1, Wt, 1024, 2048);
  gcn_gemm_bf16<<<dim3(391, 16), 256, 0, stream>>>(Xb, Wt, b1, S, NN, 1024, 2048);
  hipMemsetAsync(H, 0, (size_t)NN * 2048 * 4, stream);
  gcn_spmm<<<NE, 128, 0, stream>>>(S, ei, ev, H, 2048);

  // ---------------- layer 2: 2048 -> 1024 (ReLU on input)
  cvt_bf16_relu<<<4096, 256, 0, stream>>>(H, Xb, 50000LL * 2048, 1);
  cvt_w_transpose<<<4096, 256, 0, stream>>>(W2, Wt, 2048, 1024);
  gcn_gemm_bf16<<<dim3(391, 8), 256, 0, stream>>>(Xb, Wt, b2, S, NN, 2048, 1024);
  hipMemsetAsync(H, 0, (size_t)NN * 1024 * 4, stream);
  gcn_spmm<<<NE, 128, 0, stream>>>(S, ei, ev, H, 1024);

  // ---------------- layer 3: 1024 -> 512 (ReLU on input, no output activation)
  cvt_bf16_relu<<<4096, 256, 0, stream>>>(H, Xb, 50000LL * 1024, 1);
  cvt_w_transpose<<<4096, 256, 0, stream>>>(W3, Wt, 1024, 512);
  gcn_gemm_bf16<<<dim3(391, 4), 256, 0, stream>>>(Xb, Wt, b3, S, NN, 1024, 512);
  hipMemsetAsync(out, 0, (size_t)NN * 512 * 4, stream);
  gcn_spmm<<<NE, 128, 0, stream>>>(S, ei, ev, out, 512);
}
